// CoAttention_66554813219295
// MI455X (gfx1250) — matrix operand order repaired and checked by reading, not compiled, not run on hardware
//
#include <hip/hip_runtime.h>
#include <hip/hip_bf16.h>

typedef __attribute__((ext_vector_type(16))) __bf16 v16bf;
typedef __attribute__((ext_vector_type(8)))  __bf16 v8bf;
typedef __attribute__((ext_vector_type(8)))  float  v8f;
typedef __attribute__((ext_vector_type(4)))  float  v4f;
typedef __attribute__((ext_vector_type(4)))  unsigned int u32x4;
typedef __attribute__((ext_vector_type(8)))  int i32x8;
typedef __attribute__((ext_vector_type(4)))  int i32x4;

#define DEV static __device__ __forceinline__

#define DMODEL 512
#define NSEQ   2048
#define NBATCH 16

// ---------------------------------------------------------------------------
// WMMA wrapper: D = A(16x32 bf16) * B(32x16 bf16) + C(16x16 f32)
// ---------------------------------------------------------------------------
DEV v8f wmma_bf16(v16bf a, v16bf b, v8f c) {
  return __builtin_amdgcn_wmma_f32_16x16x32_bf16(
      false, a, false, b, (short)0, c, false, false);
}

// ---------------------------------------------------------------------------
// Fragment loaders (contiguous K = half*16 + e for A/B, wave32)
// ---------------------------------------------------------------------------
// A 16x32 bf16: lane<16 row=lane; elem e -> K = half*16 + e (sequential)
DEV v16bf load_a_bf16(const __bf16* base, int row0, int k0, int ld) {
  int lane = threadIdx.x & 31;
  int row  = lane & 15, half = lane >> 4;
  const __bf16* p = base + (size_t)(row0 + row) * ld + k0 + half * 16;
  v8bf lo = *(const v8bf*)p;          // K = half*16 + 0..7
  v8bf hi = *(const v8bf*)(p + 8);    // K = half*16 + 8..15
  v16bf f;
#pragma unroll
  for (int e = 0; e < 8; ++e) { f[e] = lo[e]; f[e + 8] = hi[e]; }
  return f;
}

// B 32x16: lane<16 col=lane; elem e -> K = half*16 + e (sequential).
// "nk": B[k][n] stored src[n][k] row-major -> per-lane contiguous 16 elements.
DEV v16bf load_b_nk_bf16(const __bf16* src, int n0, int k0, int ld) {
  int lane = threadIdx.x & 31;
  int col  = lane & 15, half = lane >> 4;
  const __bf16* p = src + (size_t)(n0 + col) * ld + k0 + half * 16;
  v8bf lo = *(const v8bf*)p;
  v8bf hi = *(const v8bf*)(p + 8);
  v16bf f;
#pragma unroll
  for (int e = 0; e < 8; ++e) { f[e] = lo[e]; f[e + 8] = hi[e]; }
  return f;
}

// B fragment out of an LDS-staged n-major tile: 64 rows (n) x 32 (K), ld=32.
// Pointer stays in the LDS address space -> ds_load_b128 x2.
DEV v16bf load_b_lds_nk(const __bf16* t, int n_off) {
  int lane = threadIdx.x & 31;
  int col  = lane & 15, half = lane >> 4;
  const __bf16* p = t + (size_t)(n_off + col) * 32 + half * 16;
  v8bf lo = *(const v8bf*)p;
  v8bf hi = *(const v8bf*)(p + 8);
  v16bf f;
#pragma unroll
  for (int e = 0; e < 8; ++e) { f[e] = lo[e]; f[e + 8] = hi[e]; }
  return f;
}

// C/D 16x16 f32: lane<16 -> N=lane, M=r; lane>=16 -> N=lane-16, M=8+r
DEV void store_c_f32(float* C, int m0, int n0, int ld, const v8f& acc) {
  int lane = threadIdx.x & 31;
  int cl = lane & 15, ch = lane >> 4;
#pragma unroll
  for (int r = 0; r < 8; ++r)
    C[(size_t)(m0 + ch * 8 + r) * ld + n0 + cl] = acc[r];
}

DEV void store_c_bf16(__bf16* C, int m0, int n0, int ld, const v8f& acc) {
  int lane = threadIdx.x & 31;
  int cl = lane & 15, ch = lane >> 4;
#pragma unroll
  for (int r = 0; r < 8; ++r)
    C[(size_t)(m0 + ch * 8 + r) * ld + n0 + cl] = (__bf16)acc[r];
}

// transposed bf16 store: Ct[n][m]; per lane the 8 M-values are contiguous -> b128
DEV void store_c_bf16_T(__bf16* Ct, int m0, int n0, int ldt, const v8f& acc) {
  int lane = threadIdx.x & 31;
  int cl = lane & 15, ch = lane >> 4;
  v8bf o;
#pragma unroll
  for (int r = 0; r < 8; ++r) o[r] = (__bf16)acc[r];
  *(v8bf*)(Ct + (size_t)(n0 + cl) * ldt + m0 + ch * 8) = o;
}

// ---------------------------------------------------------------------------
// Tensor Data Mover: 2D bf16 tile (tile_x contiguous elems, tile_y rows,
// row pitch stride_elems) -> LDS at lds_off. D# per CDNA5 ISA §8.
// ---------------------------------------------------------------------------
DEV void tdm_load_tile_bf16(unsigned lds_off, const void* gsrc,
                            unsigned tile_x, unsigned tile_y, unsigned stride_elems) {
  unsigned long long ga = (unsigned long long)gsrc;
  u32x4 g0;
  g0[0] = 1u;                                   // count=1, user mode
  g0[1] = lds_off;                              // lds_addr (bytes)
  g0[2] = (unsigned)ga;                         // global_addr[31:0]
  g0[3] = (unsigned)(ga >> 32) | 0x80000000u;   // global_addr[56:32] | type=2<<30
  i32x8 g1;
  g1[0] = 1 << 16;                              // data_size=1 -> 2 bytes/elem
  g1[1] = (int)((tile_x & 0xFFFFu) << 16);      // tensor_dim0 = tile_x  @bit48
  g1[2] = (int)((tile_x >> 16) | ((tile_y & 0xFFFFu) << 16)); // tensor_dim1
  g1[3] = (int)((tile_y >> 16) | (tile_x << 16));             // tile_dim0 @bit112
  g1[4] = (int)tile_y;                          // tile_dim1 (tile_dim2=0)
  g1[5] = (int)stride_elems;                    // tensor_dim0_stride[31:0]
  g1[6] = 0;
  g1[7] = 0;
  i32x4 gz = {0, 0, 0, 0};
#if __has_include(<hip/amd_detail/amd_gfx1250_TDM.h>)
  i32x8 gz8 = {0, 0, 0, 0, 0, 0, 0, 0};
  __builtin_amdgcn_tensor_load_to_lds(g0, g1, gz, gz, gz8, 0);
#else
  __builtin_amdgcn_tensor_load_to_lds(g0, g1, gz, gz, 0);
#endif
}

// ---------------------------------------------------------------------------
// Unified TDM-staged GEMM (first kernel in file so the disasm snippet shows
// its inner loop):  C[M x N](f32) = A[M x K](bf16) @ B, B given n-major
// (Bnk[n][k], row pitch ldb). 64x32 B tile staged via TDM, double-buffered
// in LDS, shared by all 8 waves; each wave computes a 32x64 C block
// (8 WMMAs / K-step).  grid(N/64, M/256, NBATCH) x 256
// ---------------------------------------------------------------------------
__global__ void k_tdm_gemm(const __bf16* __restrict__ Abase, int lda, unsigned long long a_batch,
                           const __bf16* __restrict__ Bbase, int ldb, unsigned long long b_batch,
                           float* __restrict__ Cbase, int ldc, unsigned long long c_batch,
                           int K) {
  const int b    = blockIdx.z;
  const int wave = threadIdx.x >> 5;
  const int m0   = (blockIdx.y * 8 + wave) * 32;
  const int n0   = blockIdx.x * 64;
  const __bf16* A    = Abase + (size_t)b * a_batch;
  const __bf16* Bsrc = Bbase + (size_t)b * b_batch + (size_t)n0 * ldb;
  float* C = Cbase + (size_t)b * c_batch;

  __shared__ __bf16 Bt[2][64 * 32];            // only LDS object -> offsets 0 / 4096
  const bool w0 = (threadIdx.x < 32);
  v8f acc0[4] = {}, acc1[4] = {};

  // The TDM writes Bt invisibly to the optimizer; give Bt a store it cannot
  // prove dead (sentinel never matches, but that is unprovable), so the LDS
  // fragment loads below neither fold to undef nor get deleted. Keeping the
  // pointer un-laundered preserves addrspace(3) inference -> ds_load_b128.
  if (blockIdx.x == 0xFFFFFFFFu) Bt[0][threadIdx.x] = (__bf16)1.0f;

  if (w0) tdm_load_tile_bf16(0u, Bsrc, 32u, 64u, (unsigned)ldb);

  const int NT = K / 32;
  for (int kk = 0; kk < NT; ++kk) {
    const int k   = kk * 32;
    const int buf = kk & 1;
    if (w0) __builtin_amdgcn_s_wait_tensorcnt(0);
    __syncthreads();                                     // tile[buf] visible to all
    if (w0 && (kk + 1) < NT)
      tdm_load_tile_bf16((unsigned)((buf ^ 1) * 64 * 32 * 2),
                         Bsrc + (k + 32), 32u, 64u, (unsigned)ldb);
    // prefetch the A stream two K-steps ahead (global_prefetch_b8)
    if (k + 64 < K)
      __builtin_prefetch(&A[(size_t)(m0 + (threadIdx.x & 31)) * lda + k + 64], 0, 1);
    const __bf16* tb = &Bt[buf][0];
    v16bf a0 = load_a_bf16(A, m0,      k, lda);
    v16bf a1 = load_a_bf16(A, m0 + 16, k, lda);
#pragma unroll
    for (int j = 0; j < 4; ++j) {
      v16bf bm = load_b_lds_nk(tb, j * 16);
      acc0[j] = wmma_bf16(a0, bm, acc0[j]);
      acc1[j] = wmma_bf16(a1, bm, acc1[j]);
    }
  }
#pragma unroll
  for (int j = 0; j < 4; ++j) {
    store_c_f32(C, m0,      n0 + j * 16, ldc, acc0[j]);
    store_c_f32(C, m0 + 16, n0 + j * 16, ldc, acc1[j]);
  }
}

// ---------------------------------------------------------------------------
// 0a) f32 -> bf16 streaming convert (8 elems/thread)
// ---------------------------------------------------------------------------
__global__ void k_cvt(const float* __restrict__ src, __bf16* __restrict__ dst, int n8) {
  int i = blockIdx.x * 256 + threadIdx.x;
  if (i >= n8) return;
  const v4f* s = (const v4f*)src;
  v4f a = s[2 * (size_t)i], b = s[2 * (size_t)i + 1];
  v8bf o;
#pragma unroll
  for (int e = 0; e < 4; ++e) { o[e] = (__bf16)a[e]; o[e + 4] = (__bf16)b[e]; }
  ((v8bf*)dst)[i] = o;
}

// ---------------------------------------------------------------------------
// 0b) transpose-convert: src f32 [NSEQ][DMODEL] -> dst bf16 [DMODEL][NSEQ]
//     grid(16, 64, 16) x (32,8)
// ---------------------------------------------------------------------------
__global__ void k_cvt_T(const float* __restrict__ src, __bf16* __restrict__ dst) {
  const int b  = blockIdx.z;
  const int d0 = blockIdx.x * 32;
  const int n0 = blockIdx.y * 32;
  __shared__ float t[32][33];
  const float* S = src + (size_t)b * NSEQ * DMODEL;
  __bf16*      T = dst + (size_t)b * NSEQ * DMODEL;
#pragma unroll
  for (int rr = 0; rr < 4; ++rr) {
    int nl = threadIdx.y + rr * 8;
    t[nl][threadIdx.x] = S[(size_t)(n0 + nl) * DMODEL + d0 + threadIdx.x];
  }
  __syncthreads();
#pragma unroll
  for (int rr = 0; rr < 4; ++rr) {
    int dl = threadIdx.y + rr * 8;
    T[(size_t)(d0 + dl) * NSEQ + n0 + threadIdx.x] = (__bf16)t[threadIdx.x][dl];
  }
}

// ---------------------------------------------------------------------------
// 1) Qb = x1b @ Wqb^T; also stores Qt (= Qb transposed).  grid(8,16,16) x 256
// ---------------------------------------------------------------------------
__global__ void k_qgemm(const __bf16* __restrict__ x1b, const __bf16* __restrict__ Wqb,
                        __bf16* __restrict__ Qb, __bf16* __restrict__ Qt) {
  const int b    = blockIdx.z;
  const int wave = threadIdx.x >> 5;
  const int m0   = (blockIdx.y * 8 + wave) * 16;
  const int n0   = blockIdx.x * 64;
  const __bf16* A = x1b + (size_t)b * NSEQ * DMODEL;
  v8f acc[4] = {};
  for (int k = 0; k < DMODEL; k += 32) {
    v16bf a = load_a_bf16(A, m0, k, DMODEL);
#pragma unroll
    for (int j = 0; j < 4; ++j) {
      v16bf bm = load_b_nk_bf16(Wqb, n0 + j * 16, k, DMODEL);
      acc[j] = wmma_bf16(a, bm, acc[j]);
    }
  }
  __bf16* C  = Qb + (size_t)b * NSEQ * DMODEL;
  __bf16* CT = Qt + (size_t)b * NSEQ * DMODEL;
#pragma unroll
  for (int j = 0; j < 4; ++j) {
    store_c_bf16  (C,  m0, n0 + j * 16, DMODEL, acc[j]);
    store_c_bf16_T(CT, m0, n0 + j * 16, NSEQ,   acc[j]);
  }
}

// ---------------------------------------------------------------------------
// 3a) row softmax stats: one block per (b,n)
// ---------------------------------------------------------------------------
__global__ void k_rowstats(const float* __restrict__ L,
                           float* __restrict__ rmax, float* __restrict__ rsum) {
  const int bn = blockIdx.x;
  const float* row = L + (size_t)bn * NSEQ;
  float m = -3.0e38f, s = 0.f;
  for (int j = threadIdx.x; j < NSEQ; j += 256) {
    float x  = row[j];
    float nm = fmaxf(m, x);
    s = s * __expf(m - nm) + __expf(x - nm);
    m = nm;
  }
  __shared__ float sm[256], ss[256];
  sm[threadIdx.x] = m; ss[threadIdx.x] = s;
  __syncthreads();
  for (int o = 128; o > 0; o >>= 1) {
    if ((int)threadIdx.x < o) {
      float m2 = sm[threadIdx.x + o], s2 = ss[threadIdx.x + o];
      float M  = fmaxf(sm[threadIdx.x], m2);
      ss[threadIdx.x] = ss[threadIdx.x] * __expf(sm[threadIdx.x] - M) + s2 * __expf(m2 - M);
      sm[threadIdx.x] = M;
    }
    __syncthreads();
  }
  if (threadIdx.x == 0) { rmax[bn] = sm[0]; rsum[bn] = ss[0]; }
}

// ---------------------------------------------------------------------------
// 3b) column softmax stats: thread-per-column, coalesced scans
// ---------------------------------------------------------------------------
__global__ void k_colstats(const float* __restrict__ L,
                           float* __restrict__ cmax, float* __restrict__ csum) {
  const int b = blockIdx.y;
  const int m = blockIdx.x * 256 + threadIdx.x;
  const float* base = L + (size_t)b * NSEQ * NSEQ + m;
  float mx = -3.0e38f, s = 0.f;
  for (int n = 0; n < NSEQ; ++n) {
    float x  = base[(size_t)n * NSEQ];
    float nm = fmaxf(mx, x);
    s = s * __expf(mx - nm) + __expf(x - nm);
    mx = nm;
  }
  cmax[b * NSEQ + m] = mx;
  csum[b * NSEQ + m] = s;
}

// ---------------------------------------------------------------------------
// 4) normalize: in-place L -> A_D; A_Q (=S1^T) via LDS transpose;
//    bf16 S1 and bf16 A_D^T into workspace.   grid(64,64,16) x (32,8)
// ---------------------------------------------------------------------------
__global__ void k_norm(float* __restrict__ L, float* __restrict__ AQ,
                       __bf16* __restrict__ S1b, __bf16* __restrict__ ADt,
                       const float* __restrict__ rmax, const float* __restrict__ rsum,
                       const float* __restrict__ cmax, const float* __restrict__ csum) {
  const int b  = blockIdx.z;
  const int m0 = blockIdx.x * 32;
  const int n0 = blockIdx.y * 32;
  __shared__ float t_ad[32][33];
  __shared__ float t_s1[32][33];
  const int tm = threadIdx.x;
  float* Lb = L + (size_t)b * NSEQ * NSEQ;
  const float* rm = rmax + b * NSEQ;  const float* rs = rsum + b * NSEQ;
  const float cmv = cmax[b * NSEQ + m0 + tm];
  const float csv = csum[b * NSEQ + m0 + tm];
#pragma unroll
  for (int rr = 0; rr < 4; ++rr) {
    const int nl = threadIdx.y + rr * 8;
    const int n  = n0 + nl;
    const size_t idx = (size_t)n * NSEQ + m0 + tm;
    float l  = Lb[idx];
    float ad = __expf(l - rm[n]) / rs[n];
    float s1 = __expf(l - cmv) / csv;
    Lb[idx] = ad;                                   // A_D in place
    S1b[(size_t)b * NSEQ * NSEQ + idx] = (__bf16)s1;
    t_ad[nl][tm] = ad;
    t_s1[nl][tm] = s1;
  }
  __syncthreads();
  float*  AQb  = AQ  + (size_t)b * NSEQ * NSEQ;
  __bf16* ADtb = ADt + (size_t)b * NSEQ * NSEQ;
#pragma unroll
  for (int rr = 0; rr < 4; ++rr) {
    const int ml = threadIdx.y + rr * 8;
    const size_t idx = (size_t)(m0 + ml) * NSEQ + n0 + tm;
    AQb[idx]  = t_s1[tm][ml];                       // A_Q[m][n] = S1[n][m]
    ADtb[idx] = (__bf16)t_ad[tm][ml];               // A_D^T[m][n]
  }
}

// ---------------------------------------------------------------------------
// host launcher
// ---------------------------------------------------------------------------
extern "C" void kernel_launch(void* const* d_in, const int* in_sizes, int n_in,
                              void* d_out, int out_size, void* d_ws, size_t ws_size,
                              hipStream_t stream) {
  (void)in_sizes; (void)n_in; (void)out_size; (void)ws_size;
  const float* x1 = (const float*)d_in[0];
  const float* x2 = (const float*)d_in[1];
  // d_in[2] = node_mask: all-ones, unused by the reference math
  const float* Wq = (const float*)d_in[3];

  float* out = (float*)d_out;
  const size_t ND = (size_t)NBATCH * NSEQ * DMODEL;   // 16,777,216
  const size_t NN = (size_t)NBATCH * NSEQ * NSEQ;     // 67,108,864
  float* CQ = out;              // [B,N1,D]
  float* CD = out + ND;         // [B,N2,D]
  float* AD = out + 2 * ND;     // [B,N1,N2]  (holds L until k_norm)
  float* AQ = AD + NN;          // [B,N2,N1]

  char* ws = (char*)d_ws;
  __bf16* Qb  = (__bf16*)ws;    ws += ND * sizeof(__bf16);
  __bf16* Qt  = (__bf16*)ws;    ws += ND * sizeof(__bf16);
  __bf16* S1b = (__bf16*)ws;    ws += NN * sizeof(__bf16);
  __bf16* ADt = (__bf16*)ws;    ws += NN * sizeof(__bf16);
  __bf16* x1b = (__bf16*)ws;    ws += ND * sizeof(__bf16);
  __bf16* Db  = (__bf16*)ws;    ws += ND * sizeof(__bf16);
  __bf16* Dt  = (__bf16*)ws;    ws += ND * sizeof(__bf16);
  __bf16* Wqb = (__bf16*)ws;    ws += (size_t)DMODEL * DMODEL * sizeof(__bf16);
  float* rmax = (float*)ws;     ws += (size_t)NBATCH * NSEQ * sizeof(float);
  float* rsum = (float*)ws;     ws += (size_t)NBATCH * NSEQ * sizeof(float);
  float* cmax = (float*)ws;     ws += (size_t)NBATCH * NSEQ * sizeof(float);
  float* csum = (float*)ws;

  // 0) bf16 conversions (+ transposed D for the C_Q GEMM)
  k_cvt<<<dim3((int)(ND / 8 / 256)), 256, 0, stream>>>(x1, x1b, (int)(ND / 8));
  k_cvt<<<dim3((int)(ND / 8 / 256)), 256, 0, stream>>>(x2, Db, (int)(ND / 8));
  k_cvt<<<dim3(DMODEL * DMODEL / 8 / 256), 256, 0, stream>>>(Wq, Wqb, DMODEL * DMODEL / 8);
  k_cvt_T<<<dim3(DMODEL / 32, NSEQ / 32, NBATCH), dim3(32, 8), 0, stream>>>(x2, Dt);
  // 1) Q projection (also emits Q^T)
  k_qgemm<<<dim3(DMODEL / 64, NSEQ / 128, NBATCH), 256, 0, stream>>>(x1b, Wqb, Qb, Qt);
  // 2) L = Q @ D^T  -> A_D slot   (TDM-staged B; 32 rows per wave)
  k_tdm_gemm<<<dim3(NSEQ / 64, NSEQ / 256, NBATCH), 256, 0, stream>>>(
      Qb, DMODEL, ND / NBATCH, Db, DMODEL, ND / NBATCH, AD, NSEQ, NN / NBATCH, DMODEL);
  // 3) softmax statistics
  k_rowstats<<<dim3(NBATCH * NSEQ), 256, 0, stream>>>(AD, rmax, rsum);
  k_colstats<<<dim3(NSEQ / 256, NBATCH), 256, 0, stream>>>(AD, cmax, csum);
  // 4) normalize + transpose + bf16 copies
  k_norm<<<dim3(NSEQ / 32, NSEQ / 32, NBATCH), dim3(32, 8), 0, stream>>>(
      AD, AQ, S1b, ADt, rmax, rsum, cmax, csum);
  // 5) C_Q = S1 @ D   (B = D^T, n-major via Dt)
  k_tdm_gemm<<<dim3(DMODEL / 64, NSEQ / 256, NBATCH), 256, 0, stream>>>(
      S1b, NSEQ, NN / NBATCH, Dt, NSEQ, ND / NBATCH, CQ, DMODEL, ND / NBATCH, NSEQ);
  // 6) C_D = A_D^T @ Q (B = Q^T, n-major via Qt)
  k_tdm_gemm<<<dim3(DMODEL / 64, NSEQ / 256, NBATCH), 256, 0, stream>>>(
      ADt, NSEQ, NN / NBATCH, Qt, NSEQ, ND / NBATCH, CD, DMODEL, ND / NBATCH, NSEQ);
}
